// Attention_19645180412111
// MI455X (gfx1250) — matrix operand order repaired and checked by reading, not compiled, not run on hardware
//
#include <hip/hip_runtime.h>

// ---------------------------------------------------------------------------
// Types for CDNA5 WMMA (wave32): v_wmma_f32_16x16x32_bf16
// ---------------------------------------------------------------------------
typedef __attribute__((ext_vector_type(16))) __bf16 v16bf;
typedef __attribute__((ext_vector_type(8)))  __bf16 v8bf;
typedef __attribute__((ext_vector_type(8)))  float  v8f;

constexpr int kDim   = 768;
constexpr int kHeads = 12;
constexpr int kHd    = 64;
constexpr int kSeq   = 4096;
constexpr int kBatch = 2;
// softmax scale = head_dim^-0.5 = 1/8, folded into Q at store time
constexpr float kScale = 0.125f;
// LDS row pitch (halfwords) for 64-wide tiles: 72 -> 144B rows, bank-conflict
// free for 128b lane accesses (36-dword stride, gcd(36,64)=4)
constexpr int kPitch = 72;

// ---------------------------------------------------------------------------
// fp32 -> bf16 (round to nearest even), as raw bits
// ---------------------------------------------------------------------------
__device__ __forceinline__ unsigned short f2bf(float f) {
    union { float f; unsigned u; } v; v.f = f;
    unsigned r = v.u + 0x7FFFu + ((v.u >> 16) & 1u);
    return (unsigned short)(r >> 16);
}

__device__ __forceinline__ v8f wmma_bf16(v16bf a, v16bf b, v8f c) {
    // (neg_a, A, neg_b, B, c_mod, C, reuse_a, reuse_b)
    return __builtin_amdgcn_wmma_f32_16x16x32_bf16(
        false, a, false, b, (short)0, c, false, false);
}

// A-fragment (16x32 bf16): lane = {m = lane&15, h = lane>>4}
// holds K = 16h + 0..15 contiguous (matches load_b16)
// caller passes k0 = kstep + 8*h ; loads 16 contig bf16 at kstep + 16*h
__device__ __forceinline__ v16bf load_a_frag(const unsigned short* base,
                                             int ld, int row, int k0) {
    const unsigned short* p = base + (size_t)row * ld + k0 + (k0 & 8);
    v8bf lo = *(const v8bf*)(p);
    v8bf hi = *(const v8bf*)(p + 8);
    return __builtin_shufflevector(lo, hi, 0,1,2,3,4,5,6,7,8,9,10,11,12,13,14,15);
}

// B-fragment (32x16 bf16): lane = {n = lane&15, h = lane>>4}
// holds K = 16h + 0..15 contiguous; loaded as two 16B chunks (no 32B-align req)
__device__ __forceinline__ v16bf load_b16(const unsigned short* p) {
    v8bf lo = *(const v8bf*)(p);
    v8bf hi = *(const v8bf*)(p + 8);
    return __builtin_shufflevector(lo, hi, 0,1,2,3,4,5,6,7,8,9,10,11,12,13,14,15);
}

// ---------------------------------------------------------------------------
// CDNA5 async global->LDS (ASYNCcnt-tracked); 16 bytes per lane.
// lds_addr: wave-relative LDS byte offset (low 32 bits of generic pointer).
// ---------------------------------------------------------------------------
__device__ __forceinline__ void async_b128(const unsigned short* g,
                                           unsigned int lds_addr) {
    asm volatile("global_load_async_to_lds_b128 %0, %1, off"
                 :: "v"(lds_addr), "v"((unsigned long long)(size_t)g)
                 : "memory");
}
__device__ __forceinline__ void wait_async0() {
    asm volatile("s_wait_asynccnt 0x0" ::: "memory");
}

// Stage a 64x64 bf16 tile (row-major, leading dim `ld`) into LDS with pitch 72.
// 256 threads: thread t copies rows t>>2, cols (t&3)*16 .. +15 (2 x 16B).
__device__ __forceinline__ void stage_tile_async(const unsigned short* gbase,
                                                 int ld, unsigned short* lbase,
                                                 int tid) {
    const int row = tid >> 2;
    const int col = (tid & 3) * 16;
    const unsigned short* g = gbase + (size_t)row * ld + col;
    const unsigned int l = (unsigned int)(size_t)(lbase + row * kPitch + col);
    async_b128(g,     l);
    async_b128(g + 8, l + 16);
}

// ---------------------------------------------------------------------------
// Elementwise fp32 -> bf16 cast
// ---------------------------------------------------------------------------
__global__ void cast_bf16_kernel(const float* __restrict__ in,
                                 unsigned short* __restrict__ out, int n) {
    int i = blockIdx.x * blockDim.x + threadIdx.x;
    int stride = gridDim.x * blockDim.x;
    for (; i < n; i += stride) out[i] = f2bf(in[i]);
}

// ---------------------------------------------------------------------------
// Cast + transpose: in[R][C] fp32 -> out[C][R] bf16 (LDS tile transpose)
// ---------------------------------------------------------------------------
__global__ __launch_bounds__(256)
void transpose_cast_kernel(const float* __restrict__ in,
                           unsigned short* __restrict__ out, int R, int C) {
    __shared__ float tile[32][33];
    const int c0 = blockIdx.x * 32, r0 = blockIdx.y * 32;
    const int tx = threadIdx.x, ty = threadIdx.y;   // block (32, 8)
    #pragma unroll
    for (int i = 0; i < 32; i += 8) {
        int r = r0 + ty + i, c = c0 + tx;
        tile[ty + i][tx] = (r < R && c < C) ? in[(size_t)r * C + c] : 0.0f;
    }
    __syncthreads();
    #pragma unroll
    for (int i = 0; i < 32; i += 8) {
        int c = c0 + ty + i, r = r0 + tx;
        if (c < C && r < R) out[(size_t)c * R + r] = f2bf(tile[tx][ty + i]);
    }
}

// ---------------------------------------------------------------------------
// QKV GEMM: [8192 x 768] @ [768 x 2304] + bias, scatter into Q/K/Vt (bf16)
//   Q : [bh][n][64]  (scaled by 1/8)     K : [bh][n][64]
//   Vt: [bh][64][n]  (transposed so P@V B-fragments load contiguously)
// 8 waves / block; wave tile 16(M) x 64(N); block tile 32 x 256
// ---------------------------------------------------------------------------
__global__ __launch_bounds__(256)
void gemm_qkv_kernel(const unsigned short* __restrict__ Ab,   // x bf16 [8192][768]
                     const unsigned short* __restrict__ Bt,   // Wqkv^T bf16 [2304][768]
                     const float* __restrict__ bias,          // [2304]
                     unsigned short* __restrict__ Qb,
                     unsigned short* __restrict__ Kb,
                     unsigned short* __restrict__ Vt) {
    const int lane = threadIdx.x & 31, w = threadIdx.x >> 5;
    const int half = lane >> 4, l16 = lane & 15;
    const int mbase = blockIdx.y * 32 + (w >> 2) * 16;
    const int nbase = blockIdx.x * 256 + (w & 3) * 64;

    v8f c[4] = {};
    const int arow = mbase + l16;
    for (int k = 0; k < kDim; k += 32) {
        v16bf a = load_a_frag(Ab, kDim, arow, k + 8 * half);
        #pragma unroll
        for (int f = 0; f < 4; ++f) {
            v16bf b = load_b16(Bt + (size_t)(nbase + f * 16 + l16) * kDim
                                  + k + 16 * half);
            c[f] = wmma_bf16(a, b, c[f]);
        }
    }

    // 64-col wave tile lies entirely inside one (t, head): uniform per wave
    const int t   = nbase / kDim;          // 0:q 1:k 2:v
    const int rem = nbase % kDim;
    const int h   = rem / kHd;
    #pragma unroll
    for (int f = 0; f < 4; ++f) {
        const int col = nbase + f * 16 + l16;
        const int d   = f * 16 + l16;      // nbase % 64 == 0
        const float bv = bias[col];
        #pragma unroll
        for (int r = 0; r < 8; ++r) {
            const int row = mbase + r + 8 * half;  // global M (C-layout)
            const int b = row >> 12;
            const int n = row & (kSeq - 1);
            const float val = c[f][r] + bv;
            const size_t hb = (size_t)(b * kHeads + h);
            if (t == 0)      Qb[(hb * kSeq + n) * kHd + d] = f2bf(val * kScale);
            else if (t == 1) Kb[(hb * kSeq + n) * kHd + d] = f2bf(val);
            else             Vt[(hb * kHd + d) * kSeq + n] = f2bf(val);
        }
    }
}

// ---------------------------------------------------------------------------
// Flash attention: one (b,h) per blockIdx.y, 128 queries per block
// (8 waves x 16 rows). K/V tiles double-buffered in LDS via async
// global->LDS DMA; online softmax; P staged through LDS for the
// C-layout -> A-layout transpose.
// ---------------------------------------------------------------------------
__global__ __launch_bounds__(256)
void flash_attn_kernel(const unsigned short* __restrict__ Qb,
                       const unsigned short* __restrict__ Kb,
                       const unsigned short* __restrict__ Vt,
                       const unsigned char* __restrict__ mask,  // [B][1][N][N]
                       unsigned short* __restrict__ Ob) {       // bf16 [B*N][768]
    __shared__ unsigned short Plds[8 * 16 * kPitch];   // 18 KB, per-wave P tile
    __shared__ unsigned short Ktile[2][64 * kPitch];   // 18 KB, double-buffered
    __shared__ unsigned short Vtile[2][64 * kPitch];   // 18 KB, double-buffered

    const int tid = threadIdx.x;
    const int lane = tid & 31, w = tid >> 5;
    const int half = lane >> 4, l16 = lane & 15;
    const int bh = blockIdx.y;                  // 0..23
    const int b = bh / kHeads, h = bh % kHeads;
    const int qbase = blockIdx.x * 128 + w * 16;

    const unsigned short* Qh = Qb + (size_t)bh * kSeq * kHd;
    const unsigned short* Kh = Kb + (size_t)bh * kSeq * kHd;
    const unsigned short* Vh = Vt + (size_t)bh * kHd * kSeq;
    const unsigned char*  Mh = mask + (size_t)b * kSeq * kSeq;

    // Q A-fragments for both d-steps, kept in registers for whole kv loop
    v16bf aq0 = load_a_frag(Qh, kHd, qbase + l16, 0  + 8 * half);
    v16bf aq1 = load_a_frag(Qh, kHd, qbase + l16, 32 + 8 * half);

    v8f o[4] = {};
    float mrow[8], lrow[8];
    #pragma unroll
    for (int r = 0; r < 8; ++r) { mrow[r] = -3.0e38f; lrow[r] = 0.0f; }

    unsigned short* Pw = &Plds[w * 16 * kPitch];

    // Prefetch first K/V tile into buffer 0 (ASYNCcnt-tracked DMA)
    stage_tile_async(Kh, kHd, &Ktile[0][0], tid);
    stage_tile_async(Vh, kSeq, &Vtile[0][0], tid);

    for (int it = 0; it < kSeq / 64; ++it) {
        const int kv0 = it * 64;
        const int buf = it & 1;

        // Tile `it` landed? (waits the 4 async ops issued last iteration)
        wait_async0();
        __syncthreads();

        // Prefetch tile it+1 into the other buffer while we compute
        if (it + 1 < kSeq / 64) {
            stage_tile_async(Kh + (size_t)(kv0 + 64) * kHd, kHd,
                             &Ktile[buf ^ 1][0], tid);
            stage_tile_async(Vh + (kv0 + 64), kSeq, &Vtile[buf ^ 1][0], tid);
        }

        const unsigned short* ldk = &Ktile[buf][0];
        const unsigned short* ldv = &Vtile[buf][0];

        // ---- S = (Q*scale) @ K^T : 16 x 64 per wave, 8 WMMAs ----
        v8f s[4] = {};
        #pragma unroll
        for (int f = 0; f < 4; ++f) {
            const int kvl = f * 16 + l16;
            v16bf bk0 = load_b16(ldk + kvl * kPitch + 16 * half);
            v16bf bk1 = load_b16(ldk + kvl * kPitch + 32 + 16 * half);
            s[f] = wmma_bf16(aq0, bk0, s[f]);
            s[f] = wmma_bf16(aq1, bk1, s[f]);
        }
        // ---- mask ----
        #pragma unroll
        for (int f = 0; f < 4; ++f) {
            const int kvcol = kv0 + f * 16 + l16;
            #pragma unroll
            for (int r = 0; r < 8; ++r) {
                const int qrow = qbase + r + 8 * half;
                if (Mh[(size_t)qrow * kSeq + kvcol]) s[f][r] = -3.0e38f;
            }
        }
        // ---- online softmax: rows live in VGPR idx, cols in lanes;
        //      xor 1/2/4/8 reduces within each 16-lane half ----
        float alpha[8];
        #pragma unroll
        for (int r = 0; r < 8; ++r) {
            float mx = fmaxf(fmaxf(s[0][r], s[1][r]), fmaxf(s[2][r], s[3][r]));
            mx = fmaxf(mx, __shfl_xor(mx, 1, 32));
            mx = fmaxf(mx, __shfl_xor(mx, 2, 32));
            mx = fmaxf(mx, __shfl_xor(mx, 4, 32));
            mx = fmaxf(mx, __shfl_xor(mx, 8, 32));
            const float mn = fmaxf(mrow[r], mx);
            alpha[r] = __expf(mrow[r] - mn);
            mrow[r] = mn;
        }
        // ---- P = exp(S - m), stage to LDS (bf16) ----
        #pragma unroll
        for (int f = 0; f < 4; ++f) {
            #pragma unroll
            for (int r = 0; r < 8; ++r) {
                const float p = __expf(s[f][r] - mrow[r]);
                s[f][r] = p;
                Pw[(r + 8 * half) * kPitch + f * 16 + l16] = f2bf(p);
            }
        }
        // ---- row sums + rescale running state ----
        #pragma unroll
        for (int r = 0; r < 8; ++r) {
            float rs = (s[0][r] + s[1][r]) + (s[2][r] + s[3][r]);
            rs += __shfl_xor(rs, 1, 32);
            rs += __shfl_xor(rs, 2, 32);
            rs += __shfl_xor(rs, 4, 32);
            rs += __shfl_xor(rs, 8, 32);
            lrow[r] = lrow[r] * alpha[r] + rs;
            #pragma unroll
            for (int f = 0; f < 4; ++f) o[f][r] *= alpha[r];
        }
        __syncthreads();
        // ---- P A-fragments from LDS (layout transpose via ds loads) ----
        v16bf ap0 = load_a_frag(Pw, kPitch, l16, 0  + 8 * half);
        v16bf ap1 = load_a_frag(Pw, kPitch, l16, 32 + 8 * half);
        // ---- O += P @ V  (V tile in LDS is [d][kv]) ----
        #pragma unroll
        for (int f = 0; f < 4; ++f) {
            const int dcol = f * 16 + l16;
            v16bf bv0 = load_b16(ldv + dcol * kPitch + 16 * half);
            v16bf bv1 = load_b16(ldv + dcol * kPitch + 32 + 16 * half);
            o[f] = wmma_bf16(ap0, bv0, o[f]);
            o[f] = wmma_bf16(ap1, bv1, o[f]);
        }
        __syncthreads();   // all waves done with `buf` before it is re-filled
    }

    // ---- finalize: O / l, write bf16 [B*N][768] at head column offset ----
    float inv[8];
    #pragma unroll
    for (int r = 0; r < 8; ++r) inv[r] = 1.0f / lrow[r];
    #pragma unroll
    for (int f = 0; f < 4; ++f) {
        const int d = f * 16 + l16;
        #pragma unroll
        for (int r = 0; r < 8; ++r) {
            const int qrow = qbase + r + 8 * half;
            Ob[((size_t)(b * kSeq + qrow)) * kDim + h * kHd + d] = f2bf(o[f][r] * inv[r]);
        }
    }
}

// ---------------------------------------------------------------------------
// Output projection: [8192 x 768] bf16 @ [768 x 768] + bias -> fp32 d_out
// ---------------------------------------------------------------------------
__global__ __launch_bounds__(256)
void gemm_proj_kernel(const unsigned short* __restrict__ Ab,   // attn out bf16
                      const unsigned short* __restrict__ Bt,   // Wproj^T bf16 [768][768]
                      const float* __restrict__ bias,
                      float* __restrict__ out) {
    const int lane = threadIdx.x & 31, w = threadIdx.x >> 5;
    const int half = lane >> 4, l16 = lane & 15;
    const int mbase = blockIdx.y * 32 + (w >> 2) * 16;
    const int nbase = blockIdx.x * 256 + (w & 3) * 64;

    v8f c[4] = {};
    const int arow = mbase + l16;
    for (int k = 0; k < kDim; k += 32) {
        v16bf a = load_a_frag(Ab, kDim, arow, k + 8 * half);
        #pragma unroll
        for (int f = 0; f < 4; ++f) {
            v16bf b = load_b16(Bt + (size_t)(nbase + f * 16 + l16) * kDim
                                  + k + 16 * half);
            c[f] = wmma_bf16(a, b, c[f]);
        }
    }
    #pragma unroll
    for (int f = 0; f < 4; ++f) {
        const int col = nbase + f * 16 + l16;
        const float bv = bias[col];
        #pragma unroll
        for (int r = 0; r < 8; ++r) {
            const int row = mbase + r + 8 * half;
            out[(size_t)row * kDim + col] = c[f][r] + bv;
        }
    }
}

// ---------------------------------------------------------------------------
// Launch
// ---------------------------------------------------------------------------
extern "C" void kernel_launch(void* const* d_in, const int* in_sizes, int n_in,
                              void* d_out, int out_size, void* d_ws, size_t ws_size,
                              hipStream_t stream) {
    (void)in_sizes; (void)n_in; (void)out_size; (void)ws_size;

    const float* x     = (const float*)d_in[0];   // [2,4096,768]
    const float* Wqkv  = (const float*)d_in[1];   // [768,2304]
    const float* bqkv  = (const float*)d_in[2];   // [2304]
    const float* Wproj = (const float*)d_in[3];   // [768,768]
    const float* bproj = (const float*)d_in[4];   // [768]
    const unsigned char* mask = (const unsigned char*)d_in[5]; // [2,1,4096,4096] bool
    float* out = (float*)d_out;

    const size_t nTok = (size_t)kBatch * kSeq;            // 8192
    char* ws = (char*)d_ws;
    size_t off = 0;
    auto carve = [&](size_t bytes) {
        char* p = ws + off;
        off += (bytes + 255) & ~(size_t)255;
        return p;
    };
    unsigned short* xb     = (unsigned short*)carve(nTok * kDim * 2);               // 12.6 MB
    unsigned short* Wqkvt  = (unsigned short*)carve((size_t)3 * kDim * kDim * 2);   //  3.5 MB
    unsigned short* Wprojt = (unsigned short*)carve((size_t)kDim * kDim * 2);       //  1.2 MB
    unsigned short* Qb     = (unsigned short*)carve(nTok * kDim * 2);               // 12.6 MB
    unsigned short* Kb     = (unsigned short*)carve(nTok * kDim * 2);               // 12.6 MB
    unsigned short* Vt     = (unsigned short*)carve(nTok * kDim * 2);               // 12.6 MB
    unsigned short* Ob     = (unsigned short*)carve(nTok * kDim * 2);               // 12.6 MB

    // 1) casts / weight transposes
    cast_bf16_kernel<<<2048, 256, 0, stream>>>(x, xb, (int)(nTok * kDim));
    transpose_cast_kernel<<<dim3(3 * kDim / 32, kDim / 32), dim3(32, 8), 0, stream>>>(
        Wqkv, Wqkvt, kDim, 3 * kDim);
    transpose_cast_kernel<<<dim3(kDim / 32, kDim / 32), dim3(32, 8), 0, stream>>>(
        Wproj, Wprojt, kDim, kDim);

    // 2) QKV projection (WMMA bf16), scatter to Q(scaled)/K/V^T
    gemm_qkv_kernel<<<dim3(3 * kDim / 256, (int)(nTok / 32)), 256, 0, stream>>>(
        xb, Wqkvt, bqkv, Qb, Kb, Vt);

    // 3) flash attention (WMMA bf16, async LDS double-buffered K/V)
    flash_attn_kernel<<<dim3(kSeq / 128, kBatch * kHeads), 256, 0, stream>>>(
        Qb, Kb, Vt, mask, Ob);

    // 4) output projection -> fp32 d_out
    gemm_proj_kernel<<<dim3(kDim / 256, (int)(nTok / 32)), 256, 0, stream>>>(
        Ob, Wprojt, bproj, out);
}
